// AdaptiveNoisingModule_7696581394520
// MI455X (gfx1250) — matrix operand order repaired and checked by reading, not compiled, not run on hardware
//
#include <hip/hip_runtime.h>
#include <cstdint>
#include <cstddef>

// Problem constants (from reference)
#define NROWS 8192
#define MROWS 16384
#define DDIM  512
#define KNN   10

#define OUT_NOISE_BASE ((size_t)NROWS * DDIM)          // 4,194,304
#define OUT_KNN_BASE   ((size_t)2 * NROWS * DDIM)      // 8,388,608 floats

// Workspace layout (float offsets)
#define WS_M2   0                    // 16384 floats
#define WS_X2   16384                // 8192 floats
#define WS_DSIG 24576                // 8192 floats
#define WS_NN   32768                // 8192 ints
#define WS_MM   40960                // 2 uints (min bits, max bits)

typedef __attribute__((ext_vector_type(16))) __bf16 v16bf;
typedef __attribute__((ext_vector_type(8)))  float  v8f;

#define WMMA_BF16(A, B, C) \
  __builtin_amdgcn_wmma_f32_16x16x32_bf16(false, (A), false, (B), (short)0, (C), false, false)

// ---------------------------------------------------------------------------
// Kernel 0: exact f32 squared norms of memory rows (m2) and feature rows (x2),
//           plus init of the global min/max atomic cells.
// One wave (32 lanes) per row, 8 rows per 256-thread block.
// ---------------------------------------------------------------------------
__global__ __launch_bounds__(256) void norms_init_kernel(
    const float* __restrict__ feat, const float* __restrict__ mem,
    float* __restrict__ ws)
{
  if (blockIdx.x == 0 && threadIdx.x == 0) {
    unsigned* mm = (unsigned*)(ws + WS_MM);
    mm[0] = 0x7F7FFFFFu;  // +FLT_MAX bits (min accumulator)
    mm[1] = 0u;           // 0.0f bits (max accumulator; dsig >= 0)
  }
  int row  = blockIdx.x * 8 + (threadIdx.x >> 5);
  int lane = threadIdx.x & 31;
  if (row >= MROWS + NROWS) return;
  const float* src = (row < MROWS) ? (mem + (size_t)row * DDIM)
                                   : (feat + (size_t)(row - MROWS) * DDIM);
  const float4* p = (const float4*)src + lane * 4;  // 16 floats per lane
  float s = 0.0f;
#pragma unroll
  for (int i = 0; i < 4; ++i) {
    float4 t = p[i];
    s += t.x * t.x + t.y * t.y + t.z * t.z + t.w * t.w;
  }
#pragma unroll
  for (int off = 16; off >= 1; off >>= 1) s += __shfl_down(s, off, 32);
  if (lane == 0) {
    if (row < MROWS) ws[WS_M2 + row] = s;
    else             ws[WS_X2 + (row - MROWS)] = s;
  }
}

// ---------------------------------------------------------------------------
// Kernel 0b: split-convert the memory bank to bf16 hi/lo planes, staged in the
// (currently dead) influence+noise region of d_out:
//   hi plane: bf16[MROWS*DDIM] at byte 0
//   lo plane: bf16[MROWS*DDIM] at byte 16 MB
// Total 32 MB == first 2*N*D floats of d_out; the knn region (written by
// kernel 1) starts after, and finalize overwrites the planes last.
// ---------------------------------------------------------------------------
__global__ __launch_bounds__(256) void bank_convert_kernel(
    const float* __restrict__ mem, __bf16* __restrict__ bank)
{
  size_t i = ((size_t)blockIdx.x * 256 + threadIdx.x) * 4;
  float4 t = *(const float4*)(mem + i);
  __bf16 h0 = (__bf16)t.x, h1 = (__bf16)t.y, h2 = (__bf16)t.z, h3 = (__bf16)t.w;
  __bf16* hi = bank + i;
  __bf16* lo = bank + (size_t)MROWS * DDIM + i;
  hi[0] = h0; hi[1] = h1; hi[2] = h2; hi[3] = h3;
  lo[0] = (__bf16)(t.x - (float)h0);
  lo[1] = (__bf16)(t.y - (float)h1);
  lo[2] = (__bf16)(t.z - (float)h2);
  lo[3] = (__bf16)(t.w - (float)h3);
}

// ---------------------------------------------------------------------------
// Kernel 1: fused distance GEMM (split-bf16 WMMA) + per-row top-10.
// Block: 256 threads (8 waves), covers 32 feature rows (2 WMMA row-tiles).
// Wave w handles col-tiles w, w+8, ... (128 tiles of 16 memory rows each).
// ---------------------------------------------------------------------------
__global__ __launch_bounds__(256) void knn_wmma_kernel(
    const float* __restrict__ feat, const __bf16* bank,
    float* __restrict__ ws, float* out)
{
  // LDS layout:
  //   Ah: 0      .. 32767   (2 rt * 16 kc * 32 lanes * 16 bf16)  A-matrix hi
  //   Al: 32768  .. 65535                                         A-matrix lo
  //   T : 65536  .. 81919   (8 waves * 32 rows * 16 cols f32)     transpose buf
  //   x2: 81920  .. 82047   (32 f32)
  // Merge phase (after barrier) aliases the dead Ah region:
  //   MD: 0      .. 10239   (32 rows * 8 waves * 10 f32)
  //   MI: 10240  .. 20479   (ints)
  __shared__ __align__(32) char smem[82048];
  __bf16* Ah  = (__bf16*)smem;
  __bf16* Al  = (__bf16*)(smem + 32768);
  float*  T   = (float*)(smem + 65536);
  float*  x2s = (float*)(smem + 81920);

  const int tid  = threadIdx.x;
  const int lane = tid & 31;
  const int wave = tid >> 5;
  const int row0 = blockIdx.x * 32;

  // ---- Stage A tile (32 rows x 512) into LDS as bf16 hi/lo in WMMA A layout.
  // A 16x32 layout used here (contiguous k): lanes 0-15 hold M=lane, K = 0..15;
  //                        lanes 16-31 hold M=lane-16, K = 16..31.
  for (int e = tid; e < 32 * DDIM; e += 256) {
    int row = e >> 9;           // 0..31
    int d   = e & 511;
    float f = feat[(size_t)(row0 + row) * DDIM + d];
    __bf16 h = (__bf16)f;
    __bf16 l = (__bf16)(f - (float)h);
    int rt = row >> 4, rl = row & 15;
    int kc = d >> 5,  o  = d & 31;
    int ln, slot;
    if (o < 16) { ln = rl;      slot = o;      }
    else        { ln = rl + 16; slot = o - 16; }
    int off = (((rt * 16 + kc) * 32) + ln) * 16 + slot;
    Ah[off] = h;
    Al[off] = l;
  }
  if (tid < 32) x2s[tid] = ws[WS_X2 + row0 + tid];
  __syncthreads();

  const v16bf* Ahv = (const v16bf*)Ah;
  const v16bf* Alv = (const v16bf*)Al;

  const int colBase = lane & 15;         // B column slot for this lane
  const int kHalf   = (lane >> 4) << 4;  // +16 d-offset for upper half lanes
  const int halfoff = (lane >> 4) << 3;  // C-layout row offset for upper lanes

  // Hoist per-row ||x||^2 values into registers (invariant over col loop).
  float x2r0[8], x2r1[8];
#pragma unroll
  for (int v = 0; v < 8; ++v) {
    x2r0[v] = x2s[halfoff + v];
    x2r1[v] = x2s[16 + halfoff + v];
  }

  // Per-lane top-10 (squared distances) for row `lane` of this block,
  // over this wave's column partition.
  float topd[KNN];
  int   topi[KNN];
#pragma unroll
  for (int k = 0; k < KNN; ++k) { topd[k] = 3.4e38f; topi[k] = 0; }

  const __bf16* lo_plane = bank + (size_t)MROWS * DDIM;

  for (int ct = wave; ct < MROWS / 16; ct += 8) {
    const int m0 = ct << 4;
    // B layout (32x16 bf16): lanes 0-15 -> N=lane, K=0..15 (consecutive d);
    //                        lanes 16-31 -> N=lane-16, K=16..31.
    // Column n of B == memory row m0+n; pre-split planes are row-major, so
    // each lane streams 16 consecutive bf16 (32 B) per k-step per plane.
    const __bf16* bh_ptr = bank     + (size_t)(m0 + colBase) * DDIM + kHalf;
    const __bf16* bl_ptr = lo_plane + (size_t)(m0 + colBase) * DDIM + kHalf;
    if (ct + 8 < MROWS / 16) {  // prefetch next tile's rows into cache
      __builtin_prefetch(bank     + (size_t)(m0 + 128 + colBase) * DDIM, 0, 1);
      __builtin_prefetch(lo_plane + (size_t)(m0 + 128 + colBase) * DDIM, 0, 1);
    }

    v8f acc0 = {};  // row-tile 0 (rows 0..15)
    v8f acc1 = {};  // row-tile 1 (rows 16..31)

    for (int kc = 0; kc < 16; ++kc) {
      v16bf bh = *(const v16bf*)(bh_ptr + kc * 32);
      v16bf bl = *(const v16bf*)(bl_ptr + kc * 32);
      int i0 = (0 * 16 + kc) * 32 + lane;
      int i1 = (1 * 16 + kc) * 32 + lane;
      v16bf ah0 = Ahv[i0], al0 = Alv[i0];
      v16bf ah1 = Ahv[i1], al1 = Alv[i1];
      // cross ~= hi*hi + hi*lo + lo*hi  (f32 accumulate)
      acc0 = WMMA_BF16(ah0, bh, acc0);
      acc0 = WMMA_BF16(ah0, bl, acc0);
      acc0 = WMMA_BF16(al0, bh, acc0);
      acc1 = WMMA_BF16(ah1, bh, acc1);
      acc1 = WMMA_BF16(ah1, bl, acc1);
      acc1 = WMMA_BF16(al1, bh, acc1);
    }

    // dist^2 = x2 + m2 - 2*cross, clamped; transpose via this wave's LDS slab
    // so each lane owns one row's 16 candidates.
    float m2v = ws[WS_M2 + m0 + colBase];
#pragma unroll
    for (int v = 0; v < 8; ++v) {
      int r0 = v + halfoff;        // C layout: VGPR v, lanes>=16 -> M=v+8
      int r1 = 16 + v + halfoff;
      float dsq0 = fmaxf(x2r0[v] + m2v - 2.0f * acc0[v], 0.0f);
      float dsq1 = fmaxf(x2r1[v] + m2v - 2.0f * acc1[v], 0.0f);
      T[(wave * 32 + r0) * 16 + colBase] = dsq0;
      T[(wave * 32 + r1) * 16 + colBase] = dsq1;
    }
    // Same-wave cross-lane LDS dependency: drain DS ops before reading back.
    asm volatile("s_wait_dscnt 0" ::: "memory");
#pragma unroll
    for (int cc = 0; cc < 16; ++cc) {
      float d = T[(wave * 32 + lane) * 16 + cc];
      if (d < topd[KNN - 1]) {
        float cd = d; int ci = m0 + cc;
#pragma unroll
        for (int j = 0; j < KNN; ++j) {
          if (cd < topd[j]) {
            float td = topd[j]; int ti = topi[j];
            topd[j] = cd; topi[j] = ci;
            cd = td; ci = ti;
          }
        }
      }
    }
  }

  // ---- Merge 8 wave-partial top-10 lists per row (alias dead A region).
  __syncthreads();
  float* MD = (float*)smem;
  int*   MI = (int*)(smem + 10240);
#pragma unroll
  for (int k = 0; k < KNN; ++k) {
    MD[(lane * 8 + wave) * KNN + k] = topd[k];
    MI[(lane * 8 + wave) * KNN + k] = topi[k];
  }
  __syncthreads();

  if (tid < 32) {
    float bd[KNN]; int bi[KNN];
#pragma unroll
    for (int k = 0; k < KNN; ++k) { bd[k] = 3.4e38f; bi[k] = 0; }
    for (int w = 0; w < 8; ++w) {
#pragma unroll
      for (int k = 0; k < KNN; ++k) {
        float d = MD[(tid * 8 + w) * KNN + k];
        if (d < bd[KNN - 1]) {
          float cd = d; int ci = MI[(tid * 8 + w) * KNN + k];
#pragma unroll
          for (int j = 0; j < KNN; ++j) {
            if (cd < bd[j]) {
              float td = bd[j]; int ti = bi[j];
              bd[j] = cd; bi[j] = ci;
              cd = td; ci = ti;
            }
          }
        }
      }
    }
    int grow = row0 + tid;
    float sum = 0.0f;
#pragma unroll
    for (int k = 0; k < KNN; ++k) {
      float dd = sqrtf(bd[k] + 1e-8f);  // reference: sqrt(clip(dsq)+EPS)
      out[OUT_KNN_BASE + (size_t)grow * KNN + k] = dd;
      sum += dd;
    }
    float dsig = sum * 0.1f;
    ws[WS_DSIG + grow] = dsig;
    ((int*)ws)[WS_NN + grow] = bi[0];
    unsigned bits = __float_as_uint(dsig);  // dsig >= 0: uint order == float order
    atomicMin((unsigned*)(ws + WS_MM), bits);
    atomicMax((unsigned*)(ws + WS_MM) + 1, bits);
  }
}

// ---------------------------------------------------------------------------
// Kernel 2: influence + noise_std epilogue. One block per feature row.
// (Overwrites the staged bf16 planes with the real influence/noise outputs.)
// ---------------------------------------------------------------------------
__global__ __launch_bounds__(256) void finalize_kernel(
    const float* __restrict__ feat, const float* __restrict__ mem,
    const float* __restrict__ isc, const float* __restrict__ dsc,
    const float* __restrict__ ws, float* __restrict__ out)
{
  __shared__ float smn[256];
  __shared__ float smx[256];
  const int row = blockIdx.x;
  const int tid = threadIdx.x;

  int   nn   = ((const int*)ws)[WS_NN + row];
  float d0   = out[OUT_KNN_BASE + (size_t)row * KNN];
  float inv  = 1.0f / (d0 + 1e-8f);
  const float* fr = feat + (size_t)row * DDIM;
  const float* mr = mem + (size_t)nn * DDIM;

  float i0 = fabsf(fr[tid]       - mr[tid])       * inv;
  float i1 = fabsf(fr[tid + 256] - mr[tid + 256]) * inv;
  out[(size_t)row * DDIM + tid]       = i0;
  out[(size_t)row * DDIM + tid + 256] = i1;

  smn[tid] = fminf(i0, i1);
  smx[tid] = fmaxf(i0, i1);
  __syncthreads();
#pragma unroll
  for (int s = 128; s > 0; s >>= 1) {
    if (tid < s) {
      smn[tid] = fminf(smn[tid], smn[tid + s]);
      smx[tid] = fmaxf(smx[tid], smx[tid + s]);
    }
    __syncthreads();
  }
  float mn = smn[0], mx = smx[0];

  float dmin = __uint_as_float(((const unsigned*)(ws + WS_MM))[0]);
  float dmax = __uint_as_float(((const unsigned*)(ws + WS_MM))[1]);
  float dnorm = (ws[WS_DSIG + row] - dmin) / (dmax - dmin + 1e-8f);
  float is = isc[0], ds = dsc[0];
  float rin = 1.0f / (mx - mn + 1e-8f);

  float c0 = is * ((i0 - mn) * rin) + ds * dnorm;
  float c1 = is * ((i1 - mn) * rin) + ds * dnorm;
  float n0 = 0.01f + 0.49f / (1.0f + expf(0.5f - c0));
  float n1 = 0.01f + 0.49f / (1.0f + expf(0.5f - c1));
  out[OUT_NOISE_BASE + (size_t)row * DDIM + tid]       = n0;
  out[OUT_NOISE_BASE + (size_t)row * DDIM + tid + 256] = n1;
}

// ---------------------------------------------------------------------------
extern "C" void kernel_launch(void* const* d_in, const int* in_sizes, int n_in,
                              void* d_out, int out_size, void* d_ws, size_t ws_size,
                              hipStream_t stream) {
  (void)in_sizes; (void)n_in; (void)out_size; (void)ws_size;
  const float* feat = (const float*)d_in[0];
  const float* mem  = (const float*)d_in[1];
  const float* isc  = (const float*)d_in[2];
  const float* dsc  = (const float*)d_in[3];
  float* out = (float*)d_out;
  float* ws  = (float*)d_ws;
  __bf16* bank = (__bf16*)d_out;  // staged bf16 planes in dead output space

  norms_init_kernel<<<(MROWS + NROWS) / 8, 256, 0, stream>>>(feat, mem, ws);
  bank_convert_kernel<<<(MROWS * DDIM) / 1024, 256, 0, stream>>>(mem, bank);
  knn_wmma_kernel<<<NROWS / 32, 256, 0, stream>>>(feat, bank, ws, out);
  finalize_kernel<<<NROWS, 256, 0, stream>>>(feat, mem, isc, dsc, ws, out);
}
